// SINGLE_Encoder_76416058130703
// MI455X (gfx1250) — compile-verified, matrix operand order checked by reading, not run on hardware
//
#include <hip/hip_runtime.h>
#include <hip/hip_bf16.h>

typedef __attribute__((ext_vector_type(16))) _Float16 v16h;
typedef __attribute__((ext_vector_type(8)))  _Float16 v8h;
typedef __attribute__((ext_vector_type(8)))  float    v8f;

#define Bsz 32
#define Nn  512
#define Ee  512
#define Hh  16
#define Dd  32
#define FFd 2048

// Fragment layout (CDNA5 ISA, 16-bit 16x32 A/B operand):
// lane group g = lane>>4; per-lane halfs 0..7  hold K = g*8  + 0..7,
//                          per-lane halfs 8..15 hold K = 16 + g*8 + 0..7.
union AF { v16h v; _Float16 h[16]; v8h q2[2]; };

__device__ __forceinline__ v8f wmma_f16(const v16h& a, const v16h& b, const v8f& c) {
  return __builtin_amdgcn_wmma_f32_16x16x32_f16(false, a, false, b, (short)0, c, false, false);
}

__device__ __forceinline__ v8h ldg8h(const _Float16* p) {
  return *(const v8h*)p;
}

// DPP lane-select immediates for xor butterflies within 8-lane groups
#define DPP8_SEL(a,b,c,d,e,f,g,h) \
  ((a)|((b)<<3)|((c)<<6)|((d)<<9)|((e)<<12)|((f)<<15)|((g)<<18)|((h)<<21))
#define SEL_X1 DPP8_SEL(1,0,3,2,5,4,7,6)
#define SEL_X2 DPP8_SEL(2,3,0,1,6,7,4,5)
#define SEL_X4 DPP8_SEL(4,5,6,7,0,1,2,3)

// max-reduce across the 16 lanes of each half-wave row group, no LDS traffic:
// xor1/2/4 via DPP8, xor8 via DPP16 row_ror:8 (rotate by half of 16-lane row).
__device__ __forceinline__ float rowmax16(float v) {
  v = fmaxf(v, __int_as_float(__builtin_amdgcn_mov_dpp8(__float_as_int(v), SEL_X1)));
  v = fmaxf(v, __int_as_float(__builtin_amdgcn_mov_dpp8(__float_as_int(v), SEL_X2)));
  v = fmaxf(v, __int_as_float(__builtin_amdgcn_mov_dpp8(__float_as_int(v), SEL_X4)));
  v = fmaxf(v, __int_as_float(__builtin_amdgcn_update_dpp(
                   0, __float_as_int(v), 0x128 /*row_ror:8*/, 0xf, 0xf, true)));
  return v;
}

// ---------------------------------------------------------------------------
// Double-buffered tiled GEMM: out[M,Nout] = act( A[M,K] * W[K,Nout] + bias )
// block = 256 threads (8 waves); tile 128(M) x 64(N), BK = 32.
// Each wave computes a 32x32 sub-tile = 2x2 WMMA tiles.
// VT: store output transposed as [b, col, n] (used to produce V^T).
// ---------------------------------------------------------------------------
template <typename TA, typename TO, bool RELU, bool VT>
__global__ __launch_bounds__(256) void gemm_wmma(
    const TA* __restrict__ A, const float* __restrict__ W,
    const float* __restrict__ bias, TO* __restrict__ out,
    int M, int K, int Nout) {
  __shared__ __align__(16) _Float16 As[2][128][48];    // row stride 96B
  __shared__ __align__(16) _Float16 Bs_t[2][64][48];   // transposed B tile [col][k]

  const int tid   = threadIdx.x;
  const int lane  = tid & 31;
  const int wid   = tid >> 5;
  const int g     = (lane >> 4) & 1;
  const int l16   = lane & 15;
  const int waveM = wid & 3;
  const int waveN = wid >> 2;
  const int mBase = blockIdx.y * 128;
  const int nBase = blockIdx.x * 64;

  const int arow = tid >> 1;
  const int acb  = (tid & 1) * 16;
  const int brow = tid >> 3;
  const int bcb  = (tid & 7) * 8;

  const TA*    aSrcBase = A + (size_t)(mBase + arow) * K + acb;
  const float* bSrcBase = W + (size_t)brow * Nout + nBase + bcb;

  v8f acc[2][2] = {};

  // prologue: stage tile 0 into buffer 0
  {
    const TA* src = aSrcBase;
#pragma unroll
    for (int i = 0; i < 16; ++i) As[0][arow][acb + i] = (_Float16)(float)src[i];
    const float* srcb = bSrcBase;
#pragma unroll
    for (int i = 0; i < 8; ++i) Bs_t[0][bcb + i][brow] = (_Float16)srcb[i];
  }
  __syncthreads();

  for (int kb = 0; kb < K; kb += 32) {
    const int cur = (kb >> 5) & 1;
    const bool hasNext = (kb + 32) < K;

    // issue next tile's global loads first (latency hidden under WMMAs)
    TA    aReg[16];
    float bReg[8];
    if (hasNext) {
      const TA* src = aSrcBase + kb + 32;
#pragma unroll
      for (int i = 0; i < 16; ++i) aReg[i] = src[i];
      const float* srcb = bSrcBase + (size_t)(kb + 32) * Nout;
#pragma unroll
      for (int i = 0; i < 8; ++i) bReg[i] = srcb[i];
      if (kb + 64 < K) {
        __builtin_prefetch(aSrcBase + kb + 64, 0, 3);
        __builtin_prefetch(bSrcBase + (size_t)(kb + 64) * Nout, 0, 3);
      }
    }

    // compute from current buffer
    AF af[2], bf[2];
#pragma unroll
    for (int ms = 0; ms < 2; ++ms) {
      const int r = waveM * 32 + ms * 16 + l16;
      af[ms].q2[0] = ldg8h(&As[cur][r][g * 8]);
      af[ms].q2[1] = ldg8h(&As[cur][r][16 + g * 8]);
    }
#pragma unroll
    for (int ns = 0; ns < 2; ++ns) {
      const int c = waveN * 32 + ns * 16 + l16;
      bf[ns].q2[0] = ldg8h(&Bs_t[cur][c][g * 8]);
      bf[ns].q2[1] = ldg8h(&Bs_t[cur][c][16 + g * 8]);
    }
#pragma unroll
    for (int ms = 0; ms < 2; ++ms)
#pragma unroll
      for (int ns = 0; ns < 2; ++ns)
        acc[ms][ns] = wmma_f16(af[ms].v, bf[ns].v, acc[ms][ns]);

    // convert + store next tile into the other buffer
    if (hasNext) {
      const int nxt = cur ^ 1;
#pragma unroll
      for (int i = 0; i < 16; ++i) As[nxt][arow][acb + i] = (_Float16)(float)aReg[i];
#pragma unroll
      for (int i = 0; i < 8; ++i) Bs_t[nxt][bcb + i][brow] = (_Float16)bReg[i];
    }
    __syncthreads();
  }

#pragma unroll
  for (int ms = 0; ms < 2; ++ms)
#pragma unroll
    for (int ns = 0; ns < 2; ++ns) {
      const int gcol = nBase + waveN * 32 + ns * 16 + l16;
      const float bv = bias ? bias[gcol] : 0.0f;
#pragma unroll
      for (int e = 0; e < 8; ++e) {
        const int grow = mBase + waveM * 32 + ms * 16 + e + 8 * g;
        float v = acc[ms][ns][e] + bv;
        if (RELU) v = fmaxf(v, 0.0f);
        if (VT) {
          const size_t idx = (size_t)(grow >> 9) * ((size_t)Ee * Nn) +
                             (size_t)gcol * Nn + (grow & (Nn - 1));
          out[idx] = (TO)v;
        } else {
          out[(size_t)grow * Nout + gcol] = (TO)v;
        }
      }
    }
}

// ---------------------------------------------------------------------------
// Flash attention with fused edge bias.
// block = 128 threads (4 waves); each wave owns 16 query rows for one (b,h).
// grid = B * H * (N/64).  V is pre-transposed: Vt[b][h*D+d][n].
// Softmax: max via DPP butterflies, running denominator via P x ones WMMA.
// ---------------------------------------------------------------------------
__global__ __launch_bounds__(128) void attn_flash(
    const _Float16* __restrict__ Q, const _Float16* __restrict__ Km,
    const _Float16* __restrict__ Vt, const float* __restrict__ edge,
    const float* __restrict__ We, const float* __restrict__ be,
    _Float16* __restrict__ O) {
  __shared__ __align__(16) _Float16 pbuf[4][16][48];

  const int blk   = blockIdx.x;
  const int qc    = blk & 7;
  const int hd    = (blk >> 3) & 15;
  const int b     = blk >> 7;
  const int wid   = threadIdx.x >> 5;
  const int lane  = threadIdx.x & 31;
  const int g     = (lane >> 4) & 1;
  const int l16   = lane & 15;
  const int qBase = qc * 64 + wid * 16;

  const float wE = We[hd];
  const float bE = be[hd];
  const float scale = 0.17677669529663687f;  // 1/sqrt(32)

  AF qf;
  {
    const _Float16* p = Q + ((size_t)(b * Nn + qBase + l16)) * Ee + hd * Dd;
    qf.q2[0] = ldg8h(p + g * 8);
    qf.q2[1] = ldg8h(p + 16 + g * 8);
  }

  AF ones;
#pragma unroll
  for (int i = 0; i < 16; ++i) ones.h[i] = (_Float16)1.0f;

  float m[8];
#pragma unroll
  for (int e = 0; e < 8; ++e) m[e] = -INFINITY;
  v8f lv = {};           // running softmax denominator (C-fragment layout)
  v8f o0 = {}, o1 = {};

  for (int kb = 0; kb < Nn; kb += 32) {
    // K^T fragments: lane = key column; d contiguous -> 2 x b128 each
    AF kf[2];
#pragma unroll
    for (int t = 0; t < 2; ++t) {
      const int key = kb + t * 16 + l16;
      const _Float16* p = Km + ((size_t)(b * Nn + key)) * Ee + hd * Dd;
      kf[t].q2[0] = ldg8h(p + g * 8);
      kf[t].q2[1] = ldg8h(p + 16 + g * 8);
    }
    v8f z = {};
    v8f s0 = wmma_f16(qf.v, kf[0].v, z);
    v8f s1 = wmma_f16(qf.v, kf[1].v, z);

    float p0[8], p1[8], corr[8];
#pragma unroll
    for (int e = 0; e < 8; ++e) {
      const int q = qBase + e + 8 * g;
      const size_t ebase = ((size_t)(b * Nn + q)) * Nn + kb;
      const float b0 = edge[ebase + l16] * wE + bE;
      const float b1 = edge[ebase + 16 + l16] * wE + bE;
      const float s0e = (s0[e] + b0) * scale;
      const float s1e = (s1[e] + b1) * scale;
      const float tm = rowmax16(fmaxf(s0e, s1e));
      const float mnew = fmaxf(m[e], tm);
      corr[e] = __expf(m[e] - mnew);
      p0[e] = __expf(s0e - mnew);
      p1[e] = __expf(s1e - mnew);
      m[e] = mnew;
      o0[e] *= corr[e];
      o1[e] *= corr[e];
    }

    // transpose P (C layout -> A layout) through per-wave LDS
#pragma unroll
    for (int e = 0; e < 8; ++e) {
      const int rp = e + 8 * g;
      pbuf[wid][rp][l16]      = (_Float16)p0[e];
      pbuf[wid][rp][16 + l16] = (_Float16)p1[e];
    }
    AF pf;
    pf.q2[0] = ldg8h(&pbuf[wid][l16][g * 8]);
    pf.q2[1] = ldg8h(&pbuf[wid][l16][16 + g * 8]);

    // running denominator: l_new = P x ones + l_old * corr  (one WMMA)
    v8f cvec;
#pragma unroll
    for (int e = 0; e < 8; ++e) cvec[e] = lv[e] * corr[e];
    lv = wmma_f16(pf.v, ones.v, cvec);

    // V fragments from V^T: lane = d column; keys contiguous -> 2 x b128 each
    AF vf[2];
#pragma unroll
    for (int t = 0; t < 2; ++t) {
      const int d = t * 16 + l16;
      const _Float16* p = Vt + ((size_t)b * Ee + hd * Dd + d) * Nn + kb;
      vf[t].q2[0] = ldg8h(p + g * 8);
      vf[t].q2[1] = ldg8h(p + 16 + g * 8);
    }
    o0 = wmma_f16(pf.v, vf[0].v, o0);
    o1 = wmma_f16(pf.v, vf[1].v, o1);
  }

#pragma unroll
  for (int e = 0; e < 8; ++e) {
    const float inv = 1.0f / lv[e];
    const int q = qBase + e + 8 * g;
    const size_t base = ((size_t)(b * Nn + q)) * Ee + hd * Dd;
    O[base + l16]      = (_Float16)(o0[e] * inv);
    O[base + 16 + l16] = (_Float16)(o1[e] * inv);
  }
}

// ---------------------------------------------------------------------------
// out = InstanceNorm_N( x + r ) * gamma + beta   (per (b, channel e))
// ---------------------------------------------------------------------------
__global__ __launch_bounds__(256) void add_instnorm(
    const float* __restrict__ x, const float* __restrict__ r,
    const float* __restrict__ gamma, const float* __restrict__ beta,
    float* __restrict__ out) {
  const int b = blockIdx.y;
  const int e = blockIdx.x * 256 + threadIdx.x;
  float s = 0.0f, sq = 0.0f;
  const size_t base = (size_t)b * Nn * Ee + e;
  for (int n = 0; n < Nn; ++n) {
    const float v = x[base + (size_t)n * Ee] + r[base + (size_t)n * Ee];
    s += v;
    sq += v * v;
  }
  const float mean = s * (1.0f / Nn);
  const float var  = sq * (1.0f / Nn) - mean * mean;
  const float rstd = rsqrtf(var + 1e-5f);
  const float ga = gamma[e], bb = beta[e];
  for (int n = 0; n < Nn; ++n) {
    const float v = x[base + (size_t)n * Ee] + r[base + (size_t)n * Ee];
    out[base + (size_t)n * Ee] = (v - mean) * rstd * ga + bb;
  }
}

// ---------------------------------------------------------------------------
extern "C" void kernel_launch(void* const* d_in, const int* in_sizes, int n_in,
                              void* d_out, int out_size, void* d_ws, size_t ws_size,
                              hipStream_t stream) {
  const float* input1 = (const float*)d_in[0];
  const float* edge   = (const float*)d_in[1];
  const float* Wq  = (const float*)d_in[2];
  const float* Wk  = (const float*)d_in[3];
  const float* Wv  = (const float*)d_in[4];
  const float* Wc  = (const float*)d_in[5];
  const float* bc  = (const float*)d_in[6];
  const float* We  = (const float*)d_in[7];
  const float* be  = (const float*)d_in[8];
  const float* g1  = (const float*)d_in[9];
  const float* b1  = (const float*)d_in[10];
  const float* W1f = (const float*)d_in[11];
  const float* b1f = (const float*)d_in[12];
  const float* W2f = (const float*)d_in[13];
  const float* b2f = (const float*)d_in[14];
  const float* g2  = (const float*)d_in[15];
  const float* b2  = (const float*)d_in[16];
  float* out = (float*)d_out;

  const int M = Bsz * Nn;  // 16384

  char* ws = (char*)d_ws;
  size_t off = 0;
  auto carve = [&](size_t bytes) -> void* {
    void* p = ws + off;
    off += (bytes + 255) & ~(size_t)255;
    return p;
  };
  _Float16* Qh    = (_Float16*)carve((size_t)M * Ee * 2);
  _Float16* Kh    = (_Float16*)carve((size_t)M * Ee * 2);
  _Float16* Vth   = (_Float16*)carve((size_t)M * Ee * 2);   // V^T [b, h*D+d, n]
  _Float16* Ah    = (_Float16*)carve((size_t)M * Ee * 2);
  float*    mh    = (float*)carve((size_t)M * Ee * 4);
  float*    out1  = (float*)carve((size_t)M * Ee * 4);
  _Float16* ffn1h = (_Float16*)carve((size_t)M * FFd * 2);
  float*    ffn2  = (float*)carve((size_t)M * Ee * 4);
  (void)ws_size; (void)in_sizes; (void)n_in; (void)out_size;

  const dim3 gE(Ee / 64, M / 128);
  const dim3 gF(FFd / 64, M / 128);
  const dim3 blk(256);

  // QKV projections (f32 in, f16 out); V written transposed
  gemm_wmma<float, _Float16, false, false><<<gE, blk, 0, stream>>>(input1, Wq, nullptr, Qh, M, Ee, Ee);
  gemm_wmma<float, _Float16, false, false><<<gE, blk, 0, stream>>>(input1, Wk, nullptr, Kh, M, Ee, Ee);
  gemm_wmma<float, _Float16, false, true ><<<gE, blk, 0, stream>>>(input1, Wv, nullptr, Vth, M, Ee, Ee);

  // attention with fused edge bias + softmax
  attn_flash<<<Bsz * Hh * (Nn / 64), 128, 0, stream>>>(Qh, Kh, Vth, edge, We, be, Ah);

  // output projection
  gemm_wmma<_Float16, float, false, false><<<gE, blk, 0, stream>>>(Ah, Wc, bc, mh, M, Ee, Ee);

  // residual + InstanceNorm #1
  add_instnorm<<<dim3(Ee / 256, Bsz), blk, 0, stream>>>(input1, mh, g1, b1, out1);

  // FFN
  gemm_wmma<float, _Float16, true, false><<<gF, blk, 0, stream>>>(out1, W1f, b1f, ffn1h, M, Ee, FFd);
  gemm_wmma<_Float16, float, false, false><<<gE, blk, 0, stream>>>(ffn1h, W2f, b2f, ffn2, M, FFd, Ee);

  // residual + InstanceNorm #2 -> final output
  add_instnorm<<<dim3(Ee / 256, Bsz), blk, 0, stream>>>(out1, ffn2, g2, b2, out);
}
